// BertSelfAttention_63221918597298
// MI455X (gfx1250) — matrix operand order repaired and checked by reading, not compiled, not run on hardware
//
#include <hip/hip_runtime.h>
#include <math.h>

// ---------------------------------------------------------------------------
// BERT self-attention with windowed relative position bias, CDNA5 (gfx1250).
// bf16 WMMA (v_wmma_f32_16x16x32_bf16) for all large GEMMs, f32 accumulate.
// Async global->LDS staging (ASYNCcnt) for GEMM tiles and the Q tile.
// ---------------------------------------------------------------------------

typedef __attribute__((ext_vector_type(16))) __bf16 v16bf;
typedef __attribute__((ext_vector_type(8)))  __bf16 v8bf;
typedef __attribute__((ext_vector_type(8)))  float  v8f;

#define D_MODEL 768
#define SEQ     1024
#define NBATCH  4
#define NHEAD   12
#define DHEAD   64
#define WINK    32
#define WINV    32

#define LDS_AS __attribute__((address_space(3)))

// LDS byte offset of a generic pointer that points into shared memory.
__device__ __forceinline__ unsigned ldsOffset(const void* p) {
    return (unsigned)(unsigned long long)(LDS_AS const char*)p;
}

// Per-lane async copy of 16B global -> LDS (tracked by ASYNCcnt, not LOADcnt).
__device__ __forceinline__ void asyncCopy16(unsigned lds_byte_off, const void* gptr) {
    asm volatile("global_load_async_to_lds_b128 %0, %1, off"
                 :: "v"(lds_byte_off), "v"(gptr) : "memory");
}

// ---------------------------------------------------------------- fp32->bf16
__global__ void cvt_bf16_kernel(const float* __restrict__ src,
                                __bf16* __restrict__ dst, int n) {
    int i = blockIdx.x * blockDim.x + threadIdx.x;
    if (i < n) dst[i] = (__bf16)src[i];
}

// ----------------------------------------------------------------- QKV GEMM
// Y[m,n] = sum_k X[m,k] * W[n,k] + bias[n];  X:[4096,768] bf16, W:[768,768] bf16
// 64x64 tile per block (8 waves). Double-buffered async LDS staging of the
// 64x32 X and W sub-tiles: global traffic = unique tile bytes (no duplication).
__global__ __launch_bounds__(256)
void qkv_gemm_kernel(const __bf16* __restrict__ X,
                     const __bf16* __restrict__ W,
                     const float*  __restrict__ bias,
                     __bf16* __restrict__ out, int transposed) {
    __shared__ __bf16 sX[2][64 * 32];   // 8 KB
    __shared__ __bf16 sW[2][64 * 32];   // 8 KB

    const int mblk = blockIdx.x * 64;
    const int nblk = blockIdx.y * 64;
    const int tid  = threadIdx.x;
    const int wave = tid >> 5;
    const int lane = tid & 31;
    const int lm = lane & 15;
    const int lh = lane >> 4;

    const int mstrip = (wave & 3) * 16;        // local m-strip
    const int nbase  = (wave >> 2) * 32;       // local n-pair base

    // stage sub-tiles for k-step `k` into buffer `buf` (one 16B copy x2/thread)
    const int srow = tid >> 2;                 // 0..63
    const int sc8  = (tid & 3) * 8;            // 0,8,16,24
    auto stage = [&](int buf, int k) {
        asyncCopy16(ldsOffset(&sX[buf][srow * 32 + sc8]),
                    X + (size_t)(mblk + srow) * D_MODEL + k + sc8);
        asyncCopy16(ldsOffset(&sW[buf][srow * 32 + sc8]),
                    W + (size_t)(nblk + srow) * D_MODEL + k + sc8);
    };

    v8f acc0 = {};
    v8f acc1 = {};

    stage(0, 0);
    const int NSTEP = D_MODEL / 32;            // 24
    for (int ks = 0; ks < NSTEP; ++ks) {
        const int cur = ks & 1;
        if (ks + 1 < NSTEP) {
            stage((ks + 1) & 1, (ks + 1) * 32);
            asm volatile("s_wait_asynccnt 0x2" ::: "memory"); // cur buf landed
        } else {
            asm volatile("s_wait_asynccnt 0x0" ::: "memory");
        }
        __syncthreads();

        // A fragment (16x32) from sX, B fragments (32x16) from sW
        union { v16bf v; v8bf h[2]; } a, b0, b1;
        const __bf16* xr = &sX[cur][(mstrip + lm) * 32];
        a.h[0] = *(const v8bf*)(xr + lh * 8);
        a.h[1] = *(const v8bf*)(xr + 16 + lh * 8);
        const __bf16* w0 = &sW[cur][(nbase + lm) * 32 + lh * 8];
        const __bf16* w1 = &sW[cur][(nbase + 16 + lm) * 32 + lh * 8];
        b0.h[0] = *(const v8bf*)(w0);
        b0.h[1] = *(const v8bf*)(w0 + 16);
        b1.h[0] = *(const v8bf*)(w1);
        b1.h[1] = *(const v8bf*)(w1 + 16);
        acc0 = __builtin_amdgcn_wmma_f32_16x16x32_bf16(false, a.v, false, b0.v,
                                                       (short)0, acc0, false, false);
        acc1 = __builtin_amdgcn_wmma_f32_16x16x32_bf16(false, a.v, false, b1.v,
                                                       (short)0, acc1, false, false);
        __syncthreads();   // protect cur buffer before it is restaged
    }

    // D layout: VGPR r -> M = r + lh*8, N = lm
    #pragma unroll
    for (int r = 0; r < 8; ++r) {
        const int m = mblk + mstrip + r + lh * 8;     // global row = b*S + s
        const int bb = m >> 10, s = m & (SEQ - 1);
        {
            const int n = nblk + nbase + lm;
            const int h = n >> 6, d = n & 63;
            const float val = acc0[r] + bias[n];
            const size_t idx = transposed
                ? ((size_t)((bb * NHEAD + h) * DHEAD + d) * SEQ + s)
                : ((size_t)((bb * NHEAD + h) * SEQ + s) * DHEAD + d);
            out[idx] = (__bf16)val;
        }
        {
            const int n = nblk + nbase + 16 + lm;
            const int h = n >> 6, d = n & 63;
            const float val = acc1[r] + bias[n];
            const size_t idx = transposed
                ? ((size_t)((bb * NHEAD + h) * DHEAD + d) * SEQ + s)
                : ((size_t)((bb * NHEAD + h) * SEQ + s) * DHEAD + d);
            out[idx] = (__bf16)val;
        }
    }
}

// ------------------------------------------------------------ fused attention
// One block per (b, h, q-tile of 32 rows). 256 threads = 8 waves.
// LDS: sQ 32x64 bf16 | sS 32x1024 f32 (128KB) | sAk/sCtx 32x66 f32 | reductions
__global__ __launch_bounds__(256)
void attn_kernel(const __bf16* __restrict__ Q,
                 const __bf16* __restrict__ K,
                 const __bf16* __restrict__ Vt,   // [B,H,DH,S]
                 const float*  __restrict__ Wrk,  // [65,64]
                 const float*  __restrict__ Wrv,  // [64,65]
                 const float*  __restrict__ mask, // [B,S]
                 float* __restrict__ out) {       // [B,S,768]
    extern __shared__ char smem[];
    __bf16* sQ  = (__bf16*)smem;                              // 4096 B
    float*  sS  = (float*)(smem + 4096);                      // 131072 B
    float*  sAk = (float*)(smem + 4096 + 131072);             // 8448 B (reused as sCtx)
    float*  sRed = (float*)(smem + 4096 + 131072 + 8448);     // 1024 B
    float*  sMax = sRed + 256;                                // 128 B
    float*  sInv = sMax + 32;                                 // 128 B

    const int b  = blockIdx.z;
    const int h  = blockIdx.y;
    const int q0 = blockIdx.x * 32;
    const int tid = threadIdx.x;
    const size_t bhS = ((size_t)b * NHEAD + h) * SEQ;   // seq-row base for Q/K

    // ---- async load Q tile (32x64 bf16) into LDS
    {
        const int qi = tid >> 3;
        const int dc = (tid & 7) * 8;
        asyncCopy16(ldsOffset(sQ + qi * DHEAD + dc),
                    Q + (bhS + q0 + qi) * DHEAD + dc);
        asm volatile("s_wait_asynccnt 0x0" ::: "memory");
    }
    __syncthreads();

    // ---- a_k[qi, r] = dot(Q[qi,:], Wrk[r,:])   (32 x 65, VALU)
    for (int p = tid; p < 32 * 65; p += 256) {
        const int qi = p / 65, r = p % 65;
        const float* wr = Wrk + r * DHEAD;
        const __bf16* qq = sQ + qi * DHEAD;
        float s = 0.0f;
        #pragma unroll 8
        for (int d = 0; d < DHEAD; ++d) s += (float)qq[d] * wr[d];
        sAk[qi * 66 + r] = s;
    }
    __syncthreads();

    const int wave = tid >> 5;
    const int lane = tid & 31;
    const int lm = lane & 15;
    const int lh = lane >> 4;

    // ---- phase 1: scores = Q K^T.  Wave owns 128 k-cols and computes BOTH
    //      m-strips so each K B-fragment is loaded once, reused by 2 WMMAs.
    {
        const int cbase = wave * 128;
        for (int nt = 0; nt < 8; ++nt) {
            const int ncol = cbase + nt * 16;
            v8f acc0 = {};
            v8f acc1 = {};
            #pragma unroll
            for (int d0 = 0; d0 < DHEAD; d0 += 32) {
                union { v16bf v; v8bf hh[2]; } a0, a1, bb;
                const __bf16* krow = K + (bhS + ncol + lm) * DHEAD + d0 + lh * 8;
                bb.hh[0] = *(const v8bf*)(krow);
                bb.hh[1] = *(const v8bf*)(krow + 16);
                const __bf16* qp0 = sQ + lm * DHEAD + d0;
                a0.hh[0] = *(const v8bf*)(qp0 + lh * 8);
                a0.hh[1] = *(const v8bf*)(qp0 + 16 + lh * 8);
                const __bf16* qp1 = sQ + (16 + lm) * DHEAD + d0;
                a1.hh[0] = *(const v8bf*)(qp1 + lh * 8);
                a1.hh[1] = *(const v8bf*)(qp1 + 16 + lh * 8);
                acc0 = __builtin_amdgcn_wmma_f32_16x16x32_bf16(
                    false, a0.v, false, bb.v, (short)0, acc0, false, false);
                acc1 = __builtin_amdgcn_wmma_f32_16x16x32_bf16(
                    false, a1.v, false, bb.v, (short)0, acc1, false, false);
            }
            const int kk = ncol + lm;
            const float mt = (1.0f - mask[(size_t)b * SEQ + kk]) * -3.4028235e38f;
            #pragma unroll
            for (int r = 0; r < 8; ++r) {
                {
                    const int qi = r + lh * 8;
                    const int rel = kk - (q0 + qi) + WINK;
                    const float bs = (rel >= 0 && rel <= 2 * WINK) ? sAk[qi * 66 + rel] : 0.0f;
                    sS[qi * SEQ + kk] = (acc0[r] + bs) * 0.125f + mt;
                }
                {
                    const int qi = 16 + r + lh * 8;
                    const int rel = kk - (q0 + qi) + WINK;
                    const float bs = (rel >= 0 && rel <= 2 * WINK) ? sAk[qi * 66 + rel] : 0.0f;
                    sS[qi * SEQ + kk] = (acc1[r] + bs) * 0.125f + mt;
                }
            }
        }
    }
    __syncthreads();

    // ---- phase 2: softmax over each of the 32 rows (8 segments of 128/row)
    {
        const int row = tid & 31, seg = tid >> 5;
        float* rowp = sS + row * SEQ;
        const int i0 = seg * 128;
        float m = -3.4028235e38f;
        for (int i = i0; i < i0 + 128; ++i) m = fmaxf(m, rowp[i]);
        sRed[row * 8 + seg] = m;
        __syncthreads();
        if (seg == 0) {
            float mm = sRed[row * 8];
            #pragma unroll
            for (int j = 1; j < 8; ++j) mm = fmaxf(mm, sRed[row * 8 + j]);
            sMax[row] = mm;
        }
        __syncthreads();
        const float mm = sMax[row];
        float s = 0.0f;
        for (int i = i0; i < i0 + 128; ++i) {
            const float e = __expf(rowp[i] - mm);
            rowp[i] = e;
            s += e;
        }
        sRed[row * 8 + seg] = s;
        __syncthreads();
        if (seg == 0) {
            float t = 0.0f;
            #pragma unroll
            for (int j = 0; j < 8; ++j) t += sRed[row * 8 + j];
            sInv[row] = 1.0f / t;
        }
        __syncthreads();
        const float inv = sInv[row];
        for (int i = i0; i < i0 + 128; ++i) rowp[i] *= inv;
    }
    __syncthreads();

    // ---- phase 3: ctx = P V (+ band @ Wrv^T).  Wave owns one 16-col d-tile
    //      and one k-half; computes BOTH m-strips (B fragment reused by 2
    //      WMMAs -> no duplicated Vt traffic), then k-halves combine via LDS.
    {
        const int ntile = (wave & 3) * 16;
        const int khalf = wave >> 2;
        const int d = ntile + lm;
        v8f acc0 = {};
        v8f acc1 = {};
        const __bf16* vbase =
            Vt + (((size_t)b * NHEAD + h) * DHEAD + d) * SEQ;
        const float* p0 = sS + lm * SEQ;
        const float* p1 = sS + (16 + lm) * SEQ;
        const int k0 = khalf * 512;
        for (int kk = k0; kk < k0 + 512; kk += 32) {
            union { v16bf v; v8bf hh[2]; } a0, a1, bb;
            const __bf16* vr = vbase + kk + lh * 8;
            bb.hh[0] = *(const v8bf*)(vr);
            bb.hh[1] = *(const v8bf*)(vr + 16);
            const int c0 = kk + lh * 8;
            const int c1 = c0 + 16;
            #pragma unroll
            for (int e = 0; e < 8; ++e) {
                a0.v[e]     = (__bf16)p0[c0 + e];
                a0.v[8 + e] = (__bf16)p0[c1 + e];
                a1.v[e]     = (__bf16)p1[c0 + e];
                a1.v[8 + e] = (__bf16)p1[c1 + e];
            }
            acc0 = __builtin_amdgcn_wmma_f32_16x16x32_bf16(
                false, a0.v, false, bb.v, (short)0, acc0, false, false);
            acc1 = __builtin_amdgcn_wmma_f32_16x16x32_bf16(
                false, a1.v, false, bb.v, (short)0, acc1, false, false);
        }

        // exchange partial sums: each wave publishes the strip it won't finish
        float* sCtx = sAk;     // 32x64 f32 fits in the (now dead) sAk region
        if (khalf == 0) {
            #pragma unroll
            for (int r = 0; r < 8; ++r) sCtx[(16 + r + lh * 8) * 64 + d] = acc1[r];
        } else {
            #pragma unroll
            for (int r = 0; r < 8; ++r) sCtx[(r + lh * 8) * 64 + d] = acc0[r];
        }
        __syncthreads();

        v8f accF;
        const int mbase = khalf ? 16 : 0;
        #pragma unroll
        for (int r = 0; r < 8; ++r) {
            accF[r] = khalf ? (acc1[r] + sCtx[(16 + r + lh * 8) * 64 + d])
                            : (acc0[r] + sCtx[(r + lh * 8) * 64 + d]);
        }

        // band contribution + output store (each wave finishes its 16 rows)
        const float* wrv = Wrv + d * (2 * WINV + 1);
        #pragma unroll
        for (int r = 0; r < 8; ++r) {
            const int qi = mbase + r + lh * 8;
            const int qg = q0 + qi;
            const float* pr = sS + qi * SEQ;
            float s = accF[r];
            const int lo = qg - WINV;
            for (int rr = 0; rr <= 2 * WINV; ++rr) {
                const int col = lo + rr;
                if (col >= 0 && col < SEQ) s += pr[col] * wrv[rr];
            }
            out[((size_t)b * SEQ + qg) * (NHEAD * DHEAD) + h * DHEAD + d] = s;
        }
    }
}

// ---------------------------------------------------------------------------
extern "C" void kernel_launch(void* const* d_in, const int* in_sizes, int n_in,
                              void* d_out, int out_size, void* d_ws, size_t ws_size,
                              hipStream_t stream) {
    const float* X    = (const float*)d_in[0];
    const float* msk  = (const float*)d_in[1];
    const float* Wq   = (const float*)d_in[2];
    const float* bq   = (const float*)d_in[3];
    const float* Wk   = (const float*)d_in[4];
    const float* bk   = (const float*)d_in[5];
    const float* Wv   = (const float*)d_in[6];
    const float* bv   = (const float*)d_in[7];
    const float* Wrk  = (const float*)d_in[8];
    const float* Wrv  = (const float*)d_in[9];
    float* out = (float*)d_out;

    const int NX = NBATCH * SEQ * D_MODEL;   // 3,145,728
    const int NW = D_MODEL * D_MODEL;        // 589,824

    __bf16* Xb  = (__bf16*)d_ws;
    __bf16* Wqb = Xb + NX;
    __bf16* Wkb = Wqb + NW;
    __bf16* Wvb = Wkb + NW;
    __bf16* Qb  = Wvb + NW;
    __bf16* Kb  = Qb + NX;
    __bf16* Vtb = Kb + NX;

    cvt_bf16_kernel<<<(NX + 255) / 256, 256, 0, stream>>>(X,  Xb,  NX);
    cvt_bf16_kernel<<<(NW + 255) / 256, 256, 0, stream>>>(Wq, Wqb, NW);
    cvt_bf16_kernel<<<(NW + 255) / 256, 256, 0, stream>>>(Wk, Wkb, NW);
    cvt_bf16_kernel<<<(NW + 255) / 256, 256, 0, stream>>>(Wv, Wvb, NW);

    dim3 gg(NBATCH * SEQ / 64, D_MODEL / 64);
    qkv_gemm_kernel<<<gg, 256, 0, stream>>>(Xb, Wqb, bq, Qb, 0);
    qkv_gemm_kernel<<<gg, 256, 0, stream>>>(Xb, Wkb, bk, Kb, 0);
    qkv_gemm_kernel<<<gg, 256, 0, stream>>>(Xb, Wvb, bv, Vtb, 1);

    dim3 ga(SEQ / 32, NHEAD, NBATCH);
    const size_t shmem = 4096 + 131072 + 8448 + 1024 + 128 + 128; // 144,896 B
    attn_kernel<<<ga, 256, shmem, stream>>>(Qb, Kb, Vtb, Wrk, Wrv, msk, out);
}
